// Tanh_RNN_68899865362507
// MI455X (gfx1250) — compile-verified, matrix operand order checked by reading, not run on hardware
//
#include <hip/hip_runtime.h>
#include <hip/hip_bf16.h>

// ---------------------------------------------------------------------------
// Tanh-RNN for MI455X (gfx1250, wave32, WMMA, async-to-LDS).
//   B=128, T=512, E=256, UNITS=256, NCLS=2
// Pipeline:
//   k_cvt_wu     : W[0], U[0] (f32 row-major) -> f16 ws. Row-major [u][e] /
//                  [u][v] IS the WMMA B-matrix layout for x@W^T and h@U^T.
//   k_embed_proj : xW = emb_table[sentence] @ W^T via f16 WMMA. W fragments
//                  register-resident; only the gathered A tile goes via LDS.
//                  Output in C-fragment lane order for the scan kernel.
//   k_rnn_scan   : 8 workgroups x 16 batch rows, 512 sequential steps.
//                  U^T fragments register-resident (64 VGPR/lane), h double-
//                  buffered in LDS, xW fragments software-pipelined with
//                  global_load_async_to_lds_b128 + s_wait_asynccnt.
//   k_head       : relu(h@W1+b1) @ W2 + b2, 2-way softmax.
// ---------------------------------------------------------------------------

#define RNN_B     128
#define RNN_T     512
#define RNN_E     256
#define RNN_U     256
#define BTILES    (RNN_B / 16)   // 8
#define TCHUNK    8

typedef _Float16 v16h __attribute__((ext_vector_type(16)));
typedef _Float16 v8h  __attribute__((ext_vector_type(8)));
typedef float    v8f  __attribute__((ext_vector_type(8)));
typedef float    v4f  __attribute__((ext_vector_type(4)));

static __device__ __forceinline__ v8f wmma_f16(v16h a, v16h b, v8f c) {
    // 8 args: (neg_a, A, neg_b, B, c_mod, C, reuse_a, reuse_b)
    return __builtin_amdgcn_wmma_f32_16x16x32_f16(
        false, a, false, b, (short)0, c, false, false);
}

static __device__ __forceinline__ v16h mk16(v8h lo, v8h hi) {
    v16h r;
#pragma unroll
    for (int j = 0; j < 8; ++j) { r[j] = lo[j]; r[j + 8] = hi[j]; }
    return r;
}

// ------------------------------------------------------------------ k_cvt_wu
// Convert W (65536 f32) and U (65536 f32) to f16 workspace copies.
__global__ __launch_bounds__(256) void k_cvt_wu(
    const float* __restrict__ W, const float* __restrict__ U,
    _Float16* __restrict__ Wh, _Float16* __restrict__ Uh) {
    int i = blockIdx.x * 256 + threadIdx.x;   // grid = 512 blocks
    if (i < RNN_U * RNN_E) Wh[i] = (_Float16)W[i];
    else                   Uh[i - RNN_U * RNN_E] = (_Float16)U[i - RNN_U * RNN_E];
}

// -------------------------------------------------------------- k_embed_proj
// grid.x = BTILES * (T/TCHUNK) = 512 workgroups, 256 threads (8 waves).
// LDS: Ah [16][256] f16 (8KB, static). W fragments live in VGPRs.
__global__ __launch_bounds__(256) void k_embed_proj(
    const int*      __restrict__ sentence,
    const float*    __restrict__ emb_table,
    const _Float16* __restrict__ Whg,     // [UNITS][E] f16 row-major
    _Float16*       __restrict__ xw)      // fragment-ordered output
{
    __shared__ __align__(16) _Float16 Ah[16 * RNN_E];

    const int tid   = threadIdx.x;
    const int lane  = tid & 31;
    const int wave  = tid >> 5;
    const int btile = blockIdx.x & (BTILES - 1);
    const int t0    = (blockIdx.x / BTILES) * TCHUNK;
    const int m     = lane & 15;          // A row / B col within tile
    const int g     = lane >> 4;          // K-group select

    // Register-resident B fragments for this wave's two N-tiles (invariant
    // across the TCHUNK timesteps this workgroup processes).
    const int n0  = wave * 2;
    const int nr0 = (n0 + 0) * 16 + m;
    const int nr1 = (n0 + 1) * 16 + m;
    v16h bf0[8], bf1[8];
#pragma unroll
    for (int ks = 0; ks < 8; ++ks) {
        const _Float16* s0 = Whg + nr0 * RNN_E + ks * 32 + g * 8;
        const _Float16* s1 = Whg + nr1 * RNN_E + ks * 32 + g * 8;
        bf0[ks] = mk16(*(const v8h*)s0, *(const v8h*)(s0 + 16));
        bf1[ks] = mk16(*(const v8h*)s1, *(const v8h*)(s1 + 16));
    }

    for (int tt = 0; tt < TCHUNK; ++tt) {
        const int t = t0 + tt;
        // Gather + convert 16 embedding rows into Ah.
        {
            const int bl  = tid >> 4;            // 0..15 local batch row
            const int ec  = (tid & 15) * 16;     // 16-col chunk
            const int b   = btile * 16 + bl;
            const int row = sentence[b * RNN_T + t];
            const float* src = emb_table + (long)row * RNN_E + ec;
#pragma unroll
            for (int q = 0; q < 4; ++q) {
                v4f e4 = *(const v4f*)(src + q * 4);
                Ah[bl * RNN_E + ec + q * 4 + 0] = (_Float16)e4.x;
                Ah[bl * RNN_E + ec + q * 4 + 1] = (_Float16)e4.y;
                Ah[bl * RNN_E + ec + q * 4 + 2] = (_Float16)e4.z;
                Ah[bl * RNN_E + ec + q * 4 + 3] = (_Float16)e4.w;
            }
        }
        __syncthreads();

        v8f acc0 = {}, acc1 = {};
#pragma unroll
        for (int ks = 0; ks < 8; ++ks) {
            const int ka = ks * 32 + g * 8;
            v16h af = mk16(*(const v8h*)&Ah[m * RNN_E + ka],
                           *(const v8h*)&Ah[m * RNN_E + ka + 16]);
            acc0 = wmma_f16(af, bf0[ks], acc0);
            acc1 = wmma_f16(af, bf1[ks], acc1);
        }

        // Store in fragment order: (((t*8+btile)*16+ntile)*32+lane)*8 halves.
        {
            long base = ((((long)t * BTILES + btile) * 16 + n0) * 32 + lane) * 8;
            v8h o0, o1;
#pragma unroll
            for (int v = 0; v < 8; ++v) {
                o0[v] = (_Float16)acc0[v];
                o1[v] = (_Float16)acc1[v];
            }
            *(v8h*)(xw + base)          = o0;
            *(v8h*)(xw + base + 32 * 8) = o1;   // next ntile stride = 256 halves
        }
        __syncthreads();   // Ah reused next tt
    }
}

// ---------------------------------------------------------------- k_rnn_scan
// grid = 8 workgroups (one per 16-row batch tile), 512 threads (16 waves).
// LDS (dynamic, 32KB): H double buffer 2x[16][256] f16 (16KB) +
//                      xW async staging double buffer 2x(512x16B) (16KB).
__global__ __launch_bounds__(512) void k_rnn_scan(
    const _Float16* __restrict__ xw,
    const _Float16* __restrict__ Uh_g,    // U[0] row-major f16 == B-matrix
    float*          __restrict__ hfinal)  // [B][UNITS] f32
{
    extern __shared__ __align__(16) _Float16 smem2[];
    _Float16* Hbuf  = smem2;                        // 2 x [16][256]
    _Float16* Stage = smem2 + 2 * 16 * RNN_U;       // 2 x 4096 halves
    const unsigned HBYTES = 2u * 16u * RNN_U * 2u;  // 16384

    const int tid   = threadIdx.x;
    const int lane  = tid & 31;
    const int wave  = tid >> 5;               // 0..15 == ntile
    const int btile = blockIdx.x;             // 0..7
    const int m     = lane & 15;
    const int g     = lane >> 4;
    const int ntile = wave;
    const int nrow  = ntile * 16 + m;         // B-matrix row in U^T operand

    // Register-resident B fragments: invariant across all 512 steps.
    v16h bfr[8];
#pragma unroll
    for (int ks = 0; ks < 8; ++ks) {
        const _Float16* src = Uh_g + nrow * RNN_U + ks * 32 + g * 8;
        bfr[ks] = mk16(*(const v8h*)src, *(const v8h*)(src + 16));
    }

    // h0 = 0 (both buffers).
    for (int i = tid * 8; i < 2 * 16 * RNN_U; i += 512 * 8) {
        v8h z = {};
        *(v8h*)&Hbuf[i] = z;
    }

    // Async-stage the t=0 xW fragment into stage buffer 0.
    // Fragment byte offset for (t,btile,ntile,lane): halves*2.
    const unsigned myFragBytes = (unsigned)tid * 16u;
    const unsigned tStrideB    = (unsigned)(BTILES * 16 * 32 * 8 * 2); // 64KB/step
    const unsigned goff0 =
        (unsigned)(((((unsigned)btile) * 16u + (unsigned)ntile) * 32u
                    + (unsigned)lane) * 8u * 2u);
    {
        unsigned ldsoff = HBYTES + myFragBytes;
        asm volatile("global_load_async_to_lds_b128 %0, %1, %2"
                     :: "v"(ldsoff), "v"(goff0), "s"(xw) : "memory");
    }
    asm volatile("s_wait_asynccnt 0x0" ::: "memory");
    __syncthreads();

    float hk[8];
#pragma unroll 1
    for (int t = 0; t < RNN_T; ++t) {
        _Float16* Hc = Hbuf + (t & 1) * (16 * RNN_U);
        _Float16* Hn = Hbuf + ((t + 1) & 1) * (16 * RNN_U);

        // Read this step's staged fragment; kick off async copy of the next.
        v8h xf = *(const v8h*)&Stage[(t & 1) * 4096 + tid * 8];
        if (t + 1 < RNN_T) {
            unsigned ldsoff = HBYTES + (unsigned)((t + 1) & 1) * 8192u + myFragBytes;
            unsigned goff   = goff0 + (unsigned)(t + 1) * tStrideB;
            asm volatile("global_load_async_to_lds_b128 %0, %1, %2"
                         :: "v"(ldsoff), "v"(goff), "s"(xw) : "memory");
        }

        v8f accA = {}, accB;
#pragma unroll
        for (int v = 0; v < 8; ++v) accB[v] = (float)xf[v];

        // Two independent 4-deep WMMA chains; B operands from registers.
#pragma unroll
        for (int ks = 0; ks < 8; ++ks) {
            const int ka = ks * 32 + g * 8;
            v16h af = mk16(*(const v8h*)&Hc[m * RNN_U + ka],
                           *(const v8h*)&Hc[m * RNN_U + ka + 16]);
            if (ks & 1) accA = wmma_f16(af, bfr[ks], accA);
            else        accB = wmma_f16(af, bfr[ks], accB);
        }

        // h_new = tanh(accA + accB); write f16 into the other buffer.
#pragma unroll
        for (int v = 0; v < 8; ++v) {
            float hv = tanhf(accA[v] + accB[v]);
            hk[v] = hv;
            const int mr = v + 8 * g;
            Hn[mr * RNN_U + ntile * 16 + m] = (_Float16)hv;
        }
        asm volatile("s_wait_asynccnt 0x0" ::: "memory");
        __syncthreads();
    }

    // Final hidden state, f32.
#pragma unroll
    for (int v = 0; v < 8; ++v) {
        const int mr = v + 8 * g;
        hfinal[(btile * 16 + mr) * RNN_U + ntile * 16 + m] = hk[v];
    }
}

// -------------------------------------------------------------------- k_head
__global__ __launch_bounds__(128) void k_head(
    const float* __restrict__ hfinal,
    const float* __restrict__ W1, const float* __restrict__ b1,
    const float* __restrict__ W2, const float* __restrict__ b2,
    float* __restrict__ out)
{
    __shared__ float sW1[RNN_U * 32];     // 32KB
    const int tid = threadIdx.x;          // one batch row per thread
    for (int i = tid; i < RNN_U * 32; i += 128) sW1[i] = W1[i];
    __syncthreads();

    float hid[32];
#pragma unroll
    for (int j = 0; j < 32; ++j) hid[j] = b1[j];
    const float* hrow = hfinal + tid * RNN_U;
    for (int e = 0; e < RNN_U; ++e) {
        const float hv = hrow[e];
#pragma unroll
        for (int j = 0; j < 32; ++j) hid[j] = fmaf(hv, sW1[e * 32 + j], hid[j]);
    }
    float l0 = b2[0], l1 = b2[1];
#pragma unroll
    for (int j = 0; j < 32; ++j) {
        const float r = fmaxf(hid[j], 0.0f);
        l0 = fmaf(r, W2[j * 2 + 0], l0);
        l1 = fmaf(r, W2[j * 2 + 1], l1);
    }
    const float mx = fmaxf(l0, l1);
    const float e0 = __expf(l0 - mx), e1 = __expf(l1 - mx);
    const float inv = 1.0f / (e0 + e1);
    out[tid * 2 + 0] = e0 * inv;
    out[tid * 2 + 1] = e1 * inv;
}

// ------------------------------------------------------------- kernel_launch
extern "C" void kernel_launch(void* const* d_in, const int* in_sizes, int n_in,
                              void* d_out, int out_size, void* d_ws, size_t ws_size,
                              hipStream_t stream) {
    const int*   sentence  = (const int*)  d_in[0];   // [B,T]
    const float* emb_table = (const float*)d_in[1];   // [VOCAB,E]
    const float* W         = (const float*)d_in[2];   // [1,U,E]
    const float* U         = (const float*)d_in[3];   // [1,U,U]
    const float* W1        = (const float*)d_in[4];   // [U,32]
    const float* b1        = (const float*)d_in[5];
    const float* W2        = (const float*)d_in[6];   // [32,2]
    const float* b2        = (const float*)d_in[7];
    float*       out       = (float*)d_out;

    // Workspace layout:
    //   [0, 32MB)              xW f16, fragment order
    //   [32MB, +128KB)         W f16 row-major
    //   [+128KB, +128KB)       U f16 row-major
    //   [+128KB)               h_final f32
    char* ws = (char*)d_ws;
    const size_t XW_BYTES = (size_t)RNN_B * RNN_T * RNN_U * 2;
    _Float16* xw_ws = (_Float16*)ws;
    _Float16* Wh_ws = (_Float16*)(ws + XW_BYTES);
    _Float16* Uh_ws = (_Float16*)(ws + XW_BYTES + (size_t)RNN_U * RNN_E * 2);
    float*    hf_ws = (float*)   (ws + XW_BYTES + (size_t)RNN_U * RNN_E * 2
                                     + (size_t)RNN_U * RNN_U * 2);

    const size_t lds2 = (size_t)(2 * 16 * RNN_U + 2 * 4096) * sizeof(_Float16); // 32KB
    (void)hipFuncSetAttribute((const void*)k_rnn_scan,
                              hipFuncAttributeMaxDynamicSharedMemorySize, (int)lds2);

    k_cvt_wu<<<2 * RNN_U * RNN_U / 256, 256, 0, stream>>>(W, U, Wh_ws, Uh_ws);
    k_embed_proj<<<BTILES * (RNN_T / TCHUNK), 256, 0, stream>>>(
        sentence, emb_table, Wh_ws, xw_ws);
    k_rnn_scan<<<BTILES, 512, lds2, stream>>>(xw_ws, Uh_ws, hf_ws);
    k_head<<<1, 128, 0, stream>>>(hf_ws, W1, b1, W2, b2, out);
}
